// AxialAttentionModule_23046794510717
// MI455X (gfx1250) — matrix operand order repaired and checked by reading, not compiled, not run on hardware
//
#include <hip/hip_runtime.h>
#include <hip/hip_bf16.h>

// Problem constants (from reference): x [B,E,C,H,W] fp32
#define B_ 2
#define E_ 10
#define C_ 64
#define H_ 128
#define W_ 256
#define S_ (H_*W_)            // 32768 spatial positions per (b,e)
#define NX (B_*E_*C_*S_)      // 41,943,040 elements

// Workspace layout (bytes): t[NX] f32 | s3[B*H*E*E] f32 | s4[B*W*E*E] f32 |
//                           Ghi/Glo/Mhi/Mlo 4096 u16 each  (~168.1 MB total)

typedef __attribute__((ext_vector_type(16))) __bf16          v16bf;
typedef __attribute__((ext_vector_type(16))) unsigned short  v16u;
typedef __attribute__((ext_vector_type(8)))  unsigned short  v8u;
typedef __attribute__((ext_vector_type(8)))  float           v8f;

__device__ __forceinline__ unsigned short f2bf(float f) {
  unsigned u = __float_as_uint(f);
  unsigned r = (u + 0x7FFFu + ((u >> 16) & 1u)) >> 16;  // round-nearest-even
  return (unsigned short)r;
}
__device__ __forceinline__ float bf2f(unsigned short b) {
  return __uint_as_float(((unsigned)b) << 16);
}
__device__ __forceinline__ float gelu_tanh(float v) {
  return 0.5f * v * (1.0f + tanhf(0.7978845608028654f * (v + 0.044715f * v * v * v)));
}

// CDNA5 16-bit WMMA fragment layouts (CDNA5 ISA 7.12.2):
//  A 16x32: lane m=lane&15; K = (i<8?i:i+8) + 8*(lane>=16)  -> two 16B runs
//  B 32x16: lane n=lane&15; K = (i<8?i:i+8) + 8*(lane>=16)  -> two 16B runs (K-major LDS)
//  C/D 16x16 f32: vgpr p -> row = p + 8*(lane>=16), col = lane&15
// Operand arrays use stride 72 u16 (144B rows): 16B-aligned rows, and the
// 36-bank row step tiles all 64 LDS banks conflict-free across a 16-lane half.
__device__ __forceinline__ v16u ld_frag_A(const unsigned short* p) {
  v8u a0 = *(const v8u*)(p);        // K run 0..7
  v8u a1 = *(const v8u*)(p + 16);   // K run 16..23
  return __builtin_shufflevector(a0, a1, 0,1,2,3,4,5,6,7,8,9,10,11,12,13,14,15);
}
__device__ __forceinline__ v16u ld_frag_B(const unsigned short* p) {
  v8u b0 = *(const v8u*)(p);
  v8u b1 = *(const v8u*)(p + 16);
  return __builtin_shufflevector(b0, b1, 0,1,2,3,4,5,6,7,8,9,10,11,12,13,14,15);
}

// ---------------------------------------------------------------------------
// Kernel 1: G = Wq^T Wk, M = Wo Wv (64x64 each), split to bf16 hi/lo.
// Also zeroes score accumulators (they are accumulated with atomics).
// ---------------------------------------------------------------------------
__global__ __launch_bounds__(256) void prep_kernel(
    const float* __restrict__ Wv, const float* __restrict__ Wk,
    const float* __restrict__ Wq, const float* __restrict__ Wo,
    unsigned short* __restrict__ Ghi, unsigned short* __restrict__ Glo,
    unsigned short* __restrict__ Mhi, unsigned short* __restrict__ Mlo,
    float* __restrict__ s3, float* __restrict__ s4) {
  int tid = threadIdx.x;
  for (int o = tid; o < 4096; o += 256) {
    int c = o >> 6, cp = o & 63;
    float accg = 0.f, accm = 0.f;
    for (int d = 0; d < 64; ++d) {
      accg += Wq[d*64 + c]  * Wk[d*64 + cp];   // G[c,cp] = sum_d Wq[d,c] Wk[d,cp]
      accm += Wo[c*64 + d]  * Wv[d*64 + cp];   // M[c,cp] = sum_d Wo[c,d] Wv[d,cp]
    }
    unsigned short gh = f2bf(accg);
    Ghi[o] = gh;  Glo[o] = f2bf(accg - bf2f(gh));
    unsigned short mh = f2bf(accm);
    Mhi[o] = mh;  Mlo[o] = f2bf(accm - bf2f(mh));
  }
  for (int o = tid; o < B_*H_*E_*E_; o += 256) s3[o] = 0.f;
  for (int o = tid; o < B_*W_*E_*E_; o += 256) s4[o] = 0.f;
}

// ---------------------------------------------------------------------------
// Kernel 2: t = G x  (per (b,e): [64x64] @ [64 x S]), bf16 hi/lo split WMMA.
// x slab is DMA'd raw into LDS via GLOBAL_LOAD_ASYNC_TO_LDS_B128 (ASYNCcnt),
// then converted LDS->LDS into K-major bf16 operand arrays.
// Grid: B*E * (S/64) blocks, 256 threads; 64c x 64s tile per workgroup.
// ---------------------------------------------------------------------------
__global__ __launch_bounds__(256) void tproj_kernel(
    const float* __restrict__ x,
    const unsigned short* __restrict__ Ghi, const unsigned short* __restrict__ Glo,
    float* __restrict__ t) {
  __shared__ unsigned short gh[64*72], gl[64*72];   // A operand [m][k]
  __shared__ unsigned short xh[64*72], xl[64*72];   // B operand K-major [w][c]
  __shared__ float xraw[64*64];                     // async staging (16 KB)
  int tid = threadIdx.x;
  int gid = blockIdx.x;
  int be  = gid >> 9;                 // / 512
  int s0  = (gid & 511) << 6;         // spatial base
  const float* xb = x + (size_t)be * (C_*S_);
  float*       tb = t + (size_t)be * (C_*S_);

  // Async DMA: 256 threads x 4 x 16B = 16 KB slab [64c][64s]
  #pragma unroll
  for (int it = 0; it < 4; ++it) {
    int f = (tid + it*256) << 2;            // float index, 16B granule
    int c = f >> 6, w = f & 63;
    const float* gp = xb + (size_t)c * S_ + s0 + w;
    unsigned loff = (unsigned)(size_t)(const void*)&xraw[f];
    asm volatile("global_load_async_to_lds_b128 %0, %1, off"
                 :: "v"(loff), "v"(gp) : "memory");
  }
  // Stage G while the DMA is in flight (8 KB, L2-resident after first WG)
  for (int o = tid; o < 4096; o += 256) {
    int r2 = o >> 6, c = o & 63;
    gh[r2*72 + c] = Ghi[o];
    gl[r2*72 + c] = Glo[o];
  }
  asm volatile("s_wait_asynccnt 0x0" ::: "memory");
  __syncthreads();
  // Convert fp32 -> bf16 hi/lo, transposed to K-major for contiguous B frags
  for (int o = tid; o < 4096; o += 256) {
    int c = o >> 6, w = o & 63;
    float v = xraw[o];
    unsigned short hb = f2bf(v);
    xh[w*72 + c] = hb;
    xl[w*72 + c] = f2bf(v - bf2f(hb));
  }
  __syncthreads();

  int lane = tid & 31, wid = tid >> 5;
  int n16 = lane & 15, hi16 = lane >> 4;
  int koffA = hi16 << 3, koffB = hi16 << 3;
  int mblk = wid & 3;                  // wave owns one M block -> reuse A frags
  int nbb  = (wid >> 2) * 2;           // two N blocks per wave
  int m = mblk*16 + n16;
  v8f acc0 = {0.f,0.f,0.f,0.f,0.f,0.f,0.f,0.f};
  v8f acc1 = {0.f,0.f,0.f,0.f,0.f,0.f,0.f,0.f};
  #pragma unroll
  for (int ks = 0; ks < 64; ks += 32) {
    v16bf ah = __builtin_bit_cast(v16bf, ld_frag_A(&gh[m*72 + ks + koffA]));
    v16bf al = __builtin_bit_cast(v16bf, ld_frag_A(&gl[m*72 + ks + koffA]));
    #pragma unroll
    for (int r = 0; r < 2; ++r) {
      int nb = (nbb + r)*16 + n16;
      v16bf bh = __builtin_bit_cast(v16bf, ld_frag_B(&xh[nb*72 + ks + koffB]));
      v16bf bl = __builtin_bit_cast(v16bf, ld_frag_B(&xl[nb*72 + ks + koffB]));
      v8f& acc = r ? acc1 : acc0;
      // fp32-class precision via bf16 hi/lo: G*X = Gh*Xh + Gh*Xl + Gl*Xh
      acc = __builtin_amdgcn_wmma_f32_16x16x32_bf16(false, ah, false, bh, (short)0, acc, false, false);
      acc = __builtin_amdgcn_wmma_f32_16x16x32_bf16(false, ah, false, bl, (short)0, acc, false, false);
      acc = __builtin_amdgcn_wmma_f32_16x16x32_bf16(false, al, false, bh, (short)0, acc, false, false);
    }
  }
  #pragma unroll
  for (int r = 0; r < 2; ++r) {
    v8f acc = r ? acc1 : acc0;
    int col = s0 + (nbb + r)*16 + n16;
    #pragma unroll
    for (int p = 0; p < 8; ++p) {
      int row = mblk*16 + p + (hi16 << 3);
      tb[(size_t)row * S_ + col] = acc[p];
    }
  }
}

// ---------------------------------------------------------------------------
// Kernel 3: fused score partials.  Per (b, h, 32-wide w tile):
//   s3[b,h,i,j] += sum_{c,w} x_i * t_j      s4[b,w,i,j] += sum_c x_i * t_j
// ---------------------------------------------------------------------------
__global__ __launch_bounds__(256) void scores_kernel(
    const float* __restrict__ x, const float* __restrict__ t,
    float* __restrict__ s3, float* __restrict__ s4) {
  __shared__ float xs[E_*C_*32];   // [j][c][w]  80 KB
  __shared__ float ts[E_*C_*32];   // 80 KB
  int tid = threadIdx.x;
  int gid = blockIdx.x;
  int b  = gid >> 10;              // H*(W/32) = 1024
  int r  = gid & 1023;
  int h  = r >> 3;
  int w0 = (r & 7) << 5;

  for (int o = tid; o < E_*C_*32; o += 256) {
    int j = o / (C_*32), c = (o >> 5) & 63, w = o & 31;
    size_t ga = ((size_t)(b*E_ + j)*C_ + c) * S_ + (size_t)h * W_ + w0 + w;
    xs[o] = x[ga];
    ts[o] = t[ga];
  }
  __syncthreads();

  if (tid < 200) {                       // 100 (i,j) pairs x 2 c-halves
    int p = tid >> 1, half = tid & 1;
    int i = p / E_, j = p % E_;
    const float* xi = xs + i*(C_*32) + half*1024;
    const float* tj = ts + j*(C_*32) + half*1024;
    float acc = 0.f;
    #pragma unroll 4
    for (int k = 0; k < 1024; ++k) acc += xi[k] * tj[k];
    atomicAdd(&s3[((size_t)(b*H_ + h)*E_ + i)*E_ + j], acc);
  }
  for (int o = tid; o < 32*E_*E_; o += 256) {
    int w = o & 31, p = o >> 5;
    int i = p / E_, j = p % E_;
    const float* xi = xs + i*(C_*32) + w;
    const float* tj = ts + j*(C_*32) + w;
    float acc = 0.f;
    #pragma unroll 4
    for (int c = 0; c < 64; ++c) acc += xi[c*32] * tj[c*32];
    atomicAdd(&s4[((size_t)(b*W_ + w0 + w)*E_ + i)*E_ + j], acc);
  }
}

// ---------------------------------------------------------------------------
// Kernel 4: in-place scaled softmax over j for both score tensors.
// ---------------------------------------------------------------------------
__global__ __launch_bounds__(256) void softmax_kernel(float* s3, float* s4) {
  int tid = blockIdx.x * blockDim.x + threadIdx.x;
  const int R3 = B_*H_*E_, R4 = B_*W_*E_;
  float* row; float sc;
  if (tid < R3)            { row = s3 + (size_t)tid*E_;        sc = 1.0f/sqrtf((float)(C_*W_)); }
  else if (tid < R3 + R4)  { row = s4 + (size_t)(tid-R3)*E_;   sc = 1.0f/sqrtf((float)(C_*H_)); }
  else return;
  float v[E_], m = -1e30f;
  #pragma unroll
  for (int j = 0; j < E_; ++j) { v[j] = row[j]*sc; m = fmaxf(m, v[j]); }
  float sum = 0.f;
  #pragma unroll
  for (int j = 0; j < E_; ++j) { v[j] = __expf(v[j]-m); sum += v[j]; }
  float inv = 1.0f/sum;
  #pragma unroll
  for (int j = 0; j < E_; ++j) row[j] = v[j]*inv;
}

// ---------------------------------------------------------------------------
// Kernel 5: z = sum_j (A3+A4) x_j ; out = gelu(M z + 2x), WMMA for M z.
// z stored K-major in LDS so B fragments are contiguous b128 loads.
// ---------------------------------------------------------------------------
__global__ __launch_bounds__(256) void final_kernel(
    const float* __restrict__ x,
    const float* __restrict__ a3, const float* __restrict__ a4,
    const unsigned short* __restrict__ Mhi, const unsigned short* __restrict__ Mlo,
    float* __restrict__ out) {
  __shared__ unsigned short zh[E_*32*72];          // [i][w][c] K-major, 45 KB
  __shared__ unsigned short zl[E_*32*72];          // 45 KB
  __shared__ unsigned short mh[64*72], ml[64*72];  // 18 KB
  __shared__ float c3[E_*E_];
  __shared__ float c4[32*E_*E_];                   // 12.8 KB
  int tid = threadIdx.x;
  int gid = blockIdx.x;
  int b  = gid >> 10;
  int r  = gid & 1023;
  int h  = r >> 3;
  int w0 = (r & 7) << 5;
  const size_t xbase = (size_t)b*E_*C_*S_ + (size_t)h*W_ + w0;

  for (int o = tid; o < 4096; o += 256) {
    int rr = o >> 6, c = o & 63;
    mh[rr*72 + c] = Mhi[o];
    ml[rr*72 + c] = Mlo[o];
  }
  if (tid < E_*E_) c3[tid] = a3[(size_t)(b*H_ + h)*E_*E_ + tid];
  for (int o = tid; o < 32*E_*E_; o += 256) {
    int w = o / (E_*E_), p = o % (E_*E_);
    c4[o] = a4[(size_t)(b*W_ + w0 + w)*E_*E_ + p];
  }
  __syncthreads();

  // z[i][c][w] = sum_j (c3[i][j] + c4[w][i][j]) * x[b,j,c,h,w0+w]  (coalesced over w)
  for (int o = tid; o < E_*C_*32; o += 256) {
    int i = o / (C_*32), c = (o >> 5) & 63, w = o & 31;
    float acc = 0.f;
    #pragma unroll
    for (int j = 0; j < E_; ++j) {
      float coef = c3[i*E_ + j] + c4[w*E_*E_ + i*E_ + j];
      acc += coef * x[xbase + ((size_t)j*C_ + c)*S_ + w];
    }
    unsigned short hb = f2bf(acc);
    zh[(i*32 + w)*72 + c] = hb;                    // K-major store
    zl[(i*32 + w)*72 + c] = f2bf(acc - bf2f(hb));
  }
  __syncthreads();

  int lane = tid & 31, wid = tid >> 5;
  int n16 = lane & 15, hi16 = lane >> 4;
  int koffA = hi16 << 3, koffB = hi16 << 3;
  for (int tile = wid; tile < 80; tile += 8) {   // 10 i x 4 mblk x 2 nblk
    int i = tile >> 3;
    int mblk = (tile >> 1) & 3;
    int nblk = tile & 1;
    int m  = mblk*16 + n16;
    int nb = i*32 + nblk*16 + n16;
    v8f acc = {0.f,0.f,0.f,0.f,0.f,0.f,0.f,0.f};
    #pragma unroll
    for (int ks = 0; ks < 64; ks += 32) {
      v16bf ah = __builtin_bit_cast(v16bf, ld_frag_A(&mh[m*72 + ks + koffA]));
      v16bf al = __builtin_bit_cast(v16bf, ld_frag_A(&ml[m*72 + ks + koffA]));
      v16bf bh = __builtin_bit_cast(v16bf, ld_frag_B(&zh[nb*72 + ks + koffB]));
      v16bf bl = __builtin_bit_cast(v16bf, ld_frag_B(&zl[nb*72 + ks + koffB]));
      acc = __builtin_amdgcn_wmma_f32_16x16x32_bf16(false, ah, false, bh, (short)0, acc, false, false);
      acc = __builtin_amdgcn_wmma_f32_16x16x32_bf16(false, ah, false, bl, (short)0, acc, false, false);
      acc = __builtin_amdgcn_wmma_f32_16x16x32_bf16(false, al, false, bh, (short)0, acc, false, false);
    }
    #pragma unroll
    for (int p = 0; p < 8; ++p) {
      int row = mblk*16 + p + (hi16 << 3);
      int w   = nblk*16 + n16;
      float xv = x[xbase + ((size_t)i*C_ + row)*S_ + w];
      float v  = acc[p] + 2.0f * xv;
      out[xbase + ((size_t)i*C_ + row)*S_ + w] = gelu_tanh(v);
    }
  }
}

// ---------------------------------------------------------------------------
extern "C" void kernel_launch(void* const* d_in, const int* in_sizes, int n_in,
                              void* d_out, int out_size, void* d_ws, size_t ws_size,
                              hipStream_t stream) {
  const float* x  = (const float*)d_in[0];
  const float* Wv = (const float*)d_in[1];
  const float* Wk = (const float*)d_in[2];
  const float* Wq = (const float*)d_in[3];
  const float* Wo = (const float*)d_in[4];
  float* out = (float*)d_out;

  char* ws = (char*)d_ws;                       // needs ~168.1 MB
  float* t  = (float*)ws;                       // NX f32
  float* s3 = (float*)(ws + (size_t)NX * 4);    // B*H*E*E
  float* s4 = s3 + B_*H_*E_*E_;                 // B*W*E*E
  unsigned short* Ghi = (unsigned short*)(s4 + B_*W_*E_*E_);
  unsigned short* Glo = Ghi + 4096;
  unsigned short* Mhi = Glo + 4096;
  unsigned short* Mlo = Mhi + 4096;

  prep_kernel   <<<1,                 256, 0, stream>>>(Wv, Wk, Wq, Wo, Ghi, Glo, Mhi, Mlo, s3, s4);
  tproj_kernel  <<<B_*E_*(S_/64),     256, 0, stream>>>(x, Ghi, Glo, t);
  scores_kernel <<<B_*H_*(W_/32),     256, 0, stream>>>(x, t, s3, s4);
  int rows = B_*H_*E_ + B_*W_*E_;
  softmax_kernel<<<(rows + 255)/256,  256, 0, stream>>>(s3, s4);
  final_kernel  <<<B_*H_*(W_/32),     256, 0, stream>>>(x, s3, s4, Mhi, Mlo, out);
}
